// BiDAF_17489106829432
// MI455X (gfx1250) — matrix operand order repaired and checked by reading, not compiled, not run on hardware
//
#include <hip/hip_runtime.h>

// ---------------------------------------------------------------------------
// Char-CNN (embed -> conv1d(D=100,F=100,K=5,pad=2) -> +bias -> maxpool over W)
// as a bf16 WMMA GEMM on gfx1250 (wave32).
//
//   A[(n,w)][kk] = emb[char(n, w+kk/100-2)][kk%100],  kk = k*100+d, Kdim=512 (pad)
//   B[kk][f]     = conv_w[f][d][k]
//   y[n,w,f]     = (A x B)[w][f];  out[n,f] = max_w y + bias[f]
//
// One wave per token: M=16 rows = the 16 char positions. 7 N-tiles of 16
// filters (100 padded to 112), 16 K-steps of 32 -> 112 v_wmma per token.
// A-fragment gather is fully branch-free: the embedding stage has 20 rows
// (row = w + k, rows 0,1,18,19 permanently zero) so window edges and K-pad
// resolve to zero rows via address selects, never EXEC masking.
// ---------------------------------------------------------------------------

typedef __attribute__((ext_vector_type(16))) __bf16          v16bf;
typedef __attribute__((ext_vector_type(16))) unsigned short  v16u;
typedef __attribute__((ext_vector_type(8)))  unsigned int    v8u;
typedef __attribute__((ext_vector_type(8)))  float           v8f;

#define NTOK   32768
#define WPOS   16
#define DDIM   100
#define FDIM   100
#define KW     5
#define NTILES 7              // 112 filter columns (100 padded)
#define KSTEPS 16             // 512 / 32
#define BPACK_ELEMS (NTILES * KSTEPS * 32 * 16)   // 57344 u16 fragment-packed B
#define WAVES_PER_BLOCK 4
#define E_ROWS 20             // rows: ri = w + k, k in 0..4 -> ri in 0..19
#define E_STRIDE 104          // u16 per row (100 + pad), 208 B, 16B-aligned

__device__ __forceinline__ unsigned short f2bf(float x) {
    unsigned u = __builtin_bit_cast(unsigned, x);
    u = u + 0x7FFFu + ((u >> 16) & 1u);       // round-to-nearest-even
    return (unsigned short)(u >> 16);
}

// ---------------------------------------------------------------------------
// Kernel 1: pack conv_w [F,D,K] f32 into bf16 B-fragments, WMMA register order.
// Flat index i = ((nt*16 + t)*32 + lane)*16 + j
//   lane<16 : B[32t + (j<8 ? j : 16+j-8)    ][nt*16 + lane]      (K = 0..7, 16..23)
//   lane>=16: B[32t + 8 + (j<8 ? j : 16+j-8)][nt*16 + lane-16]   (K = 8..15, 24..31)
// ---------------------------------------------------------------------------
__global__ void pack_b_kernel(const float* __restrict__ conv_w,
                              unsigned short* __restrict__ bpack) {
    int i = blockIdx.x * blockDim.x + threadIdx.x;
    if (i >= BPACK_ELEMS) return;
    int j    = i & 15;
    int L    = (i >> 4) & 31;
    int s    = i >> 9;            // nt*16 + t
    int t    = s & 15;
    int nt   = s >> 4;
    int half = L >> 4;
    int f    = nt * 16 + (L & 15);
    int kk   = 32 * t + ((j < 8) ? 0 : 16) + half * 8 + (j & 7);
    unsigned short v = 0;
    if (kk < DDIM * KW && f < FDIM) {
        int k = kk / DDIM;
        int d = kk - k * DDIM;
        v = f2bf(conv_w[f * (DDIM * KW) + d * KW + k]);
    }
    bpack[i] = v;
}

// ---------------------------------------------------------------------------
// Kernel 2: per-token WMMA conv + maxpool.
// Dynamic LDS: packed B (114688 B), shared by the whole block.
// Static  LDS: per-wave 20-row char-embedding stage E[20][104] bf16.
// ---------------------------------------------------------------------------
__global__ void __launch_bounds__(WAVES_PER_BLOCK * 32)
charcnn_wmma_kernel(const int*   __restrict__ char_ids,
                    const float* __restrict__ embed,
                    const float* __restrict__ conv_b,
                    const unsigned short* __restrict__ bpack,
                    float* __restrict__ out) {
    extern __shared__ __align__(16) unsigned short Bl[];          // 57344 u16
    __shared__ __align__(16) unsigned short Es[WAVES_PER_BLOCK * E_ROWS * E_STRIDE];

    const int lane = threadIdx.x & 31;
    const int wv   = threadIdx.x >> 5;
    const int w    = lane & 15;     // output row M = char position
    const int half = lane >> 4;

    // ---- stage packed B into LDS once per block (7168 x uint4) ----
    {
        uint4*       dst = (uint4*)Bl;
        const uint4* src = (const uint4*)bpack;
        for (int i = threadIdx.x; i < BPACK_ELEMS / 8; i += blockDim.x)
            dst[i] = src[i];
    }
    __syncthreads();

    unsigned short* E = Es + wv * (E_ROWS * E_STRIDE);

    // ---- zero the border rows 0,1 and 18,19 once (they stay zero) ----
    {
        unsigned* z0 = (unsigned*)(E);                        // rows 0..1
        unsigned* z1 = (unsigned*)(E + 18 * E_STRIDE);        // rows 18..19
        for (int i = lane; i < E_STRIDE; i += 32) { z0[i] = 0u; z1[i] = 0u; }
    }

    const int wave_stride = gridDim.x * WAVES_PER_BLOCK;

    for (int n = blockIdx.x * WAVES_PER_BLOCK + wv; n < NTOK; n += wave_stride) {
        // ---- stage the token's 16 char embeddings (bf16 pairs) at row c+2 ----
        const int* ids = char_ids + n * WPOS;
        #pragma unroll
        for (int c = 0; c < WPOS; ++c) {
            int id = ids[c];                        // uniform per wave
            const float2* er2 = (const float2*)(embed + id * DDIM);
            unsigned short* row = E + (c + 2) * E_STRIDE;
            for (int p = lane; p < DDIM / 2; p += 32) {
                float2 e = er2[p];
                unsigned pk = (unsigned)f2bf(e.x) | ((unsigned)f2bf(e.y) << 16);
                *(unsigned*)(row + 2 * p) = pk;
            }
        }
        asm volatile("s_wait_dscnt 0" ::: "memory");

        // ---- K-step outer, filter tiles inner; 7 live f32 accumulators ----
        v8f acc[NTILES];
        #pragma unroll
        for (int nt = 0; nt < NTILES; ++nt) acc[nt] = (v8f){};

        #pragma unroll
        for (int t = 0; t < KSTEPS; ++t) {
            // A fragment: 4 aligned sub-runs of 4 bf16 (one ds_load_b64 each).
            // Sub-run s: K base = 32t + (s<2?0:16) + 4*(s&1) + 8*half.
            // base%4==0 => d<=96, never crosses a k*100 row boundary, 8B aligned.
            v8u au;
            #pragma unroll
            for (int s = 0; s < 4; ++s) {
                const int base = 32 * t + ((s < 2) ? 0 : 16) + 4 * (s & 1);
                const int SA = base;       // half = 0
                const int SB = base + 8;   // half = 1
                const int  kA = SA / DDIM, dA = SA - (SA / DDIM) * DDIM;
                const int  kB = SB / DDIM, dB = SB - (SB / DDIM) * DDIM;
                const bool pA = (SA >= DDIM * KW);
                const bool pB = (SB >= DDIM * KW);
                const unsigned short* qa = pA ? E : (E + (w + kA) * E_STRIDE + dA);
                const unsigned short* qb = pB ? E : (E + (w + kB) * E_STRIDE + dB);
                const uint2 v = *(const uint2*)(half ? qb : qa);
                au[2 * s]     = v.x;
                au[2 * s + 1] = v.y;
            }
            const v16bf a = __builtin_bit_cast(v16bf, au);

            #pragma unroll
            for (int nt = 0; nt < NTILES; ++nt) {
                const v16u bu =
                    *(const v16u*)(Bl + ((nt * KSTEPS + t) * 32 + lane) * 16);
                acc[nt] = __builtin_amdgcn_wmma_f32_16x16x32_bf16(
                            false, a,
                            false, __builtin_bit_cast(v16bf, bu),
                            (short)0, acc[nt], false, false);
            }
        }

        // ---- max over M per tile: acc rows M=r (half 0) / M=r+8 (half 1) ----
        #pragma unroll
        for (int nt = 0; nt < NTILES; ++nt) {
            float m = acc[nt][0];
            #pragma unroll
            for (int r = 1; r < 8; ++r) m = fmaxf(m, acc[nt][r]);
            // merge halves: SWAPX16 (group-of-32, xor=0x10, and=0x1f)
            float o = __builtin_bit_cast(float,
                        __builtin_amdgcn_ds_swizzle(
                            __builtin_bit_cast(int, m), 0x401f));
            m = fmaxf(m, o);
            if (half == 0) {
                int f = nt * 16 + w;
                if (f < FDIM)
                    out[n * FDIM + f] = m + conv_b[f];
            }
        }
    }
}

extern "C" void kernel_launch(void* const* d_in, const int* in_sizes, int n_in,
                              void* d_out, int out_size, void* d_ws, size_t ws_size,
                              hipStream_t stream) {
    const int*   char_ids = (const int*)  d_in[0];
    const float* embed    = (const float*)d_in[1];
    const float* conv_w   = (const float*)d_in[2];
    const float* conv_b   = (const float*)d_in[3];
    float*       out      = (float*)d_out;
    unsigned short* bpack = (unsigned short*)d_ws;   // 114688 bytes

    pack_b_kernel<<<(BPACK_ELEMS + 255) / 256, 256, 0, stream>>>(conv_w, bpack);

    const size_t dyn_lds = (size_t)BPACK_ELEMS * sizeof(unsigned short); // 114688
    charcnn_wmma_kernel<<<512, WAVES_PER_BLOCK * 32, dyn_lds, stream>>>(
        char_ids, embed, conv_b, bpack, out);
}
